// SharedHashGridMLPEncoder_3255585210681
// MI455X (gfx1250) — compile-verified, matrix operand order checked by reading, not run on hardware
//
#include <hip/hip_runtime.h>
#include <hip/hip_bf16.h>
#include <math.h>

typedef __attribute__((ext_vector_type(16))) _Float16 v16h;
typedef __attribute__((ext_vector_type(8)))  float    v8f;

#define HASH_SIZE 32768
#define HASH_MASK 32767u
#define PRIME1 2654435761u
#define PRIME2 805459861u

// leaky(x) = max(x,0) + 0.2*min(x,0)  -> v_max_num/v_min_num/fma, no VCC traffic
__device__ __forceinline__ float leaky_act(float x) {
  return fmaxf(x, 0.f) + 0.2f * fminf(x, 0.f);
}

// One hashgrid level for one point: 8-corner trilinear gather of float4 features.
__device__ __forceinline__ void enc_level(bool dense, int res, const float* __restrict__ gl,
                                          float dx, float dy, float dz, float* acc) {
  float px = dx * (float)res, py = dy * (float)res, pz = dz * (float)res;
  float fx0 = floorf(px), fy0 = floorf(py), fz0 = floorf(pz);
  float fx = px - fx0, fy = py - fy0, fz = pz - fz0;
  int ix = (int)fx0, iy = (int)fy0, iz = (int)fz0;
  acc[0] = acc[1] = acc[2] = acc[3] = 0.f;
#pragma unroll
  for (int c = 0; c < 8; ++c) {
    const int bx = c & 1, by = (c >> 1) & 1, bz = (c >> 2) & 1;
    int cx = ix + bx; cx = cx < 0 ? 0 : (cx > res ? res : cx);
    int cy = iy + by; cy = cy < 0 ? 0 : (cy > res ? res : cy);
    int cz = iz + bz; cz = cz < 0 ? 0 : (cz > res ? res : cz);
    unsigned idx;
    if (dense) idx = (unsigned)(cx + (res + 1) * (cy + (res + 1) * cz));
    else       idx = (((unsigned)cx) ^ ((unsigned)cy * PRIME1) ^ ((unsigned)cz * PRIME2)) & HASH_MASK;
    float w = (bx ? fx : 1.f - fx) * (by ? fy : 1.f - fy) * (bz ? fz : 1.f - fz);
    const float4 g = *reinterpret_cast<const float4*>(gl + (size_t)idx * 4u);
    acc[0] += w * g.x; acc[1] += w * g.y; acc[2] += w * g.z; acc[3] += w * g.w;
  }
}

// Build 16x32 f16 A-fragment (M x K) from row-major LDS [16 rows x S f16], zero-pad K >= kreal.
// ISA layout: lanes 0-15 row M=lane hold K 0..7 (v0-3) and 16..23 (v4-7);
//             lanes 16-31 row M=lane-16 hold K 8..15 and 24..31.
__device__ __forceinline__ v16h make_a(const _Float16* __restrict__ lds, int lane, int S, int kreal) {
  const int m  = lane & 15;
  const int kb = (lane < 16) ? 0 : 8;
  v16h a;
#pragma unroll
  for (int v = 0; v < 8; ++v) {
    const int k = kb + ((v < 4) ? (2 * v) : (16 + 2 * (v - 4)));
    a[2 * v]     = (k     < kreal) ? lds[m * S + k]     : (_Float16)0.f;
    a[2 * v + 1] = (k + 1 < kreal) ? lds[m * S + k + 1] : (_Float16)0.f;
  }
  return a;
}

// Build 32x16 f16 B-fragment (K x N) from global row-major W[Kreal x Ncols] f32,
// for column tile [nofs, nofs+16). Mirrors the A-layout with N on lanes.
__device__ __forceinline__ v16h make_b(const float* __restrict__ W, int lane,
                                       int Ncols, int Kreal, int nofs) {
  const int n  = (lane & 15) + nofs;
  const int kb = (lane < 16) ? 0 : 8;
  v16h b;
#pragma unroll
  for (int v = 0; v < 8; ++v) {
    const int k = kb + ((v < 4) ? (2 * v) : (16 + 2 * (v - 4)));
    const bool nv = (n < Ncols);
    float e0 = (nv && k     < Kreal) ? W[k * Ncols + n]       : 0.f;
    float e1 = (nv && k + 1 < Kreal) ? W[(k + 1) * Ncols + n] : 0.f;
    b[2 * v]     = (_Float16)e0;
    b[2 * v + 1] = (_Float16)e1;
  }
  return b;
}

// Apply bias + leaky-relu to a 16x16 f32 D tile and store f16 to row-major LDS [16 x S].
// D layout: column n = lane&15 (+nofs), rows r (lanes 0-15) / r+8 (lanes 16-31).
__device__ __forceinline__ void store_act(_Float16* __restrict__ lds, int lane, int S,
                                          v8f d, const float* __restrict__ bias,
                                          int nofs, int Ncols) {
  const int n  = (lane & 15) + nofs;
  const int r0 = (lane < 16) ? 0 : 8;
  if (n < Ncols) {
    const float bb = bias[n];
#pragma unroll
    for (int r = 0; r < 8; ++r) {
      float h = leaky_act(d[r] + bb);
      lds[(r0 + r) * S + n] = (_Float16)h;
    }
  }
}

// One-wave kernel: pre-pack the four B-fragments (W1 tile0, W1 tile1, W2, W3)
// into f16 WMMA operand layout in workspace. Each lane owns 16 f16 = 32 bytes
// per fragment -> the hot kernel refetches each fragment with 2x global_load_b128.
__global__ __launch_bounds__(32) void pack_weights_kernel(
    const float* __restrict__ W1, const float* __restrict__ W2,
    const float* __restrict__ W3, _Float16* __restrict__ wfrag) {
  const int lane = threadIdx.x & 31;
  v16h b;
  b = make_b(W1, lane, 32, 16, 0);   *(v16h*)(wfrag + (0 * 32 + lane) * 16) = b;
  b = make_b(W1, lane, 32, 16, 16);  *(v16h*)(wfrag + (1 * 32 + lane) * 16) = b;
  b = make_b(W2, lane, 16, 32, 0);   *(v16h*)(wfrag + (2 * 32 + lane) * 16) = b;
  b = make_b(W3, lane,  8, 16, 0);   *(v16h*)(wfrag + (3 * 32 + lane) * 16) = b;
}

__global__ __launch_bounds__(256) void hashgrid_mlp_kernel(
    const float* __restrict__ dirs, const float* __restrict__ grid,
    const _Float16* __restrict__ wfrag,
    const float* __restrict__ b1, const float* __restrict__ b2,
    const float* __restrict__ b3,
    const float* __restrict__ W4, const float* __restrict__ b4,
    float* __restrict__ out, int n) {
  __shared__ _Float16 s_feat[128 * 16];  // [point][16 feats]
  __shared__ _Float16 s_h1[128 * 32];
  __shared__ _Float16 s_h2[128 * 16];
  __shared__ _Float16 s_h3[128 * 8];

  const int tid  = threadIdx.x;
  const int lane = tid & 31;
  const int wv   = tid >> 5;          // 8 waves/block, 16 points each
  const int p    = lane & 15;         // point within wave tile
  const int gp   = blockIdx.x * 128 + wv * 16 + p;

  // Fetch pre-packed weight fragments early (L2-resident, 2x b128 each);
  // latency overlaps the encode below.
  const v16h bw1a = *(const v16h*)(wfrag + (0 * 32 + lane) * 16);
  const v16h bw1b = *(const v16h*)(wfrag + (1 * 32 + lane) * 16);
  const v16h bw2  = *(const v16h*)(wfrag + (2 * 32 + lane) * 16);
  const v16h bw3  = *(const v16h*)(wfrag + (3 * 32 + lane) * 16);

  float dx = 0.f, dy = 0.f, dz = 0.f;
  if (gp < n) { dx = dirs[gp * 3 + 0]; dy = dirs[gp * 3 + 1]; dz = dirs[gp * 3 + 2]; }

  // ---- Encoding: lanes 0-15 -> levels 0,1 (dense); lanes 16-31 -> levels 2,3 (hashed)
  const bool dense = (lane < 16);
  const int  l0    = dense ? 0 : 2;
  const int  resA  = dense ? 16 : 36;
  const int  resB  = dense ? 24 : 54;

  float acc[4];
  _Float16* frow = s_feat + (wv * 16 + p) * 16;

  enc_level(dense, resA, grid + (size_t)l0 * HASH_SIZE * 4, dx, dy, dz, acc);
  frow[l0 * 4 + 0] = (_Float16)acc[0];
  frow[l0 * 4 + 1] = (_Float16)acc[1];
  frow[l0 * 4 + 2] = (_Float16)acc[2];
  frow[l0 * 4 + 3] = (_Float16)acc[3];

  enc_level(dense, resB, grid + (size_t)(l0 + 1) * HASH_SIZE * 4, dx, dy, dz, acc);
  frow[(l0 + 1) * 4 + 0] = (_Float16)acc[0];
  frow[(l0 + 1) * 4 + 1] = (_Float16)acc[1];
  frow[(l0 + 1) * 4 + 2] = (_Float16)acc[2];
  frow[(l0 + 1) * 4 + 3] = (_Float16)acc[3];

  __syncthreads();

  const v8f zero = {};

  // ---- Layer 1: [16x16 pad->32] x W1[16x32] as two 16-wide N tiles
  v16h a1  = make_a(s_feat + wv * 256, lane, 16, 16);
  v8f d1a = __builtin_amdgcn_wmma_f32_16x16x32_f16(false, a1, false, bw1a, (short)0, zero, false, false);
  v8f d1b = __builtin_amdgcn_wmma_f32_16x16x32_f16(false, a1, false, bw1b, (short)0, zero, false, false);
  store_act(s_h1 + wv * 512, lane, 32, d1a, b1, 0, 32);
  store_act(s_h1 + wv * 512, lane, 32, d1b, b1, 16, 32);
  __syncthreads();

  // ---- Layer 2: [16x32] x W2[32x16]  (exact 16x16x32 WMMA shape)
  v16h a2 = make_a(s_h1 + wv * 512, lane, 32, 32);
  v8f d2 = __builtin_amdgcn_wmma_f32_16x16x32_f16(false, a2, false, bw2, (short)0, zero, false, false);
  store_act(s_h2 + wv * 256, lane, 16, d2, b2, 0, 16);
  __syncthreads();

  // ---- Layer 3: [16x16 pad->32] x W3[16x8 pad->16]
  v16h a3 = make_a(s_h2 + wv * 256, lane, 16, 16);
  v8f d3 = __builtin_amdgcn_wmma_f32_16x16x32_f16(false, a3, false, bw3, (short)0, zero, false, false);
  store_act(s_h3 + wv * 128, lane, 8, d3, b3, 0, 8);
  __syncthreads();

  // ---- Layer 4: 8->3 + tanh, scalar per point (too small for WMMA)
  if (lane < 16) {
    const _Float16* h = s_h3 + (wv * 16 + lane) * 8;
    float o0 = b4[0], o1 = b4[1], o2 = b4[2];
#pragma unroll
    for (int k = 0; k < 8; ++k) {
      float hv = (float)h[k];
      o0 += hv * W4[k * 3 + 0];
      o1 += hv * W4[k * 3 + 1];
      o2 += hv * W4[k * 3 + 2];
    }
    const int g = blockIdx.x * 128 + wv * 16 + lane;
    if (g < n) {
      out[g * 3 + 0] = tanhf(o0);
      out[g * 3 + 1] = tanhf(o1);
      out[g * 3 + 2] = tanhf(o2);
    }
  }
}

extern "C" void kernel_launch(void* const* d_in, const int* in_sizes, int n_in,
                              void* d_out, int out_size, void* d_ws, size_t ws_size,
                              hipStream_t stream) {
  const float* dirs = (const float*)d_in[0];
  const float* grid = (const float*)d_in[1];
  const float* W1   = (const float*)d_in[2];
  const float* b1   = (const float*)d_in[3];
  const float* W2   = (const float*)d_in[4];
  const float* b2   = (const float*)d_in[5];
  const float* W3   = (const float*)d_in[6];
  const float* b3   = (const float*)d_in[7];
  const float* W4   = (const float*)d_in[8];
  const float* b4   = (const float*)d_in[9];

  _Float16* wfrag = (_Float16*)d_ws;           // 4 fragments * 32 lanes * 16 f16 = 4 KB

  const int n      = in_sizes[0] / 3;          // 2,097,152 points
  const int blocks = (n + 127) / 128;          // 128 points per 256-thread block

  hipLaunchKernelGGL(pack_weights_kernel, dim3(1), dim3(32), 0, stream,
                     W1, W2, W3, wfrag);
  hipLaunchKernelGGL(hashgrid_mlp_kernel, dim3(blocks), dim3(256), 0, stream,
                     dirs, grid, wfrag, b1, b2, b3, W4, b4,
                     (float*)d_out, n);
}
